// Attention_62397284877035
// MI455X (gfx1250) — 16-bit and 8-bit matrix operand order repaired and checked by reading, compiled, not run on hardware
//
#include <hip/hip_runtime.h>

typedef _Float16 v16h __attribute__((ext_vector_type(16)));
typedef _Float16 v8h  __attribute__((ext_vector_type(8)));
typedef _Float16 v4h  __attribute__((ext_vector_type(4)));
typedef float    v8f  __attribute__((ext_vector_type(8)));
typedef int      v8i  __attribute__((ext_vector_type(8)));
typedef int      v4ix __attribute__((ext_vector_type(4)));
typedef int      v2i  __attribute__((ext_vector_type(2)));
typedef int      v4i  __attribute__((vector_size(16)));

#define BATCH   4
#define SEQ     4096
#define DIM     64
#define M_TILE  64
#define NWAVES  4
#define NTHREADS (NWAVES * 32)
#define QSCALE  0.125f
#define QS_L2E  0.18033688011112042f
#define NEG_BIG (-1.0e30f)

#if defined(__has_builtin)
#  if __has_builtin(__builtin_amdgcn_global_load_async_to_lds_b128)
#    define FA_ASYNC 1
#  endif
#endif

static __device__ __forceinline__ void cp16(void* l, const void* g) {
#if defined(FA_ASYNC)
    __builtin_amdgcn_global_load_async_to_lds_b128(
        (__attribute__((address_space(1))) v4i*)g,
        (__attribute__((address_space(3))) v4i*)l, 0, 0);
#else
    *(v4ix*)l = *(const v4ix*)g;
#endif
}

static __device__ __forceinline__ void async_wait0() {
#if defined(FA_ASYNC)
#  if __has_builtin(__builtin_amdgcn_s_wait_asynccnt)
    __builtin_amdgcn_s_wait_asynccnt(0);
#  else
    asm volatile("s_wait_asynccnt 0" ::: "memory");
#  endif
#endif
}

static __device__ __forceinline__ v16h cat8(v8h lo, v8h hh) {
    return __builtin_shufflevector(lo, hh, 0,1,2,3,4,5,6,7,8,9,10,11,12,13,14,15);
}

__global__ __launch_bounds__(256)
void cvt_kernel(const float* __restrict__ K, const float* __restrict__ V,
                _Float16* __restrict__ Kh, unsigned char* __restrict__ V8T)
{
    __shared__ __align__(128) _Float16 tile[64][72];
    const int tid = threadIdx.x;
    const int s0  = blockIdx.x * 64;
    const int b   = blockIdx.y;
    const size_t base = ((size_t)b * SEQ + s0) * DIM;

    #pragma unroll
    for (int e = 0; e < 4; ++e) {
        const int idx4 = (e * 256 + tid) * 4;
        const int s = idx4 >> 6, d = idx4 & 63;
        const float4 kv = *(const float4*)(K + base + idx4);
        const float4 vv = *(const float4*)(V + base + idx4);
        v4h kh;
        kh[0] = (_Float16)kv.x; kh[1] = (_Float16)kv.y;
        kh[2] = (_Float16)kv.z; kh[3] = (_Float16)kv.w;
        *(v4h*)(Kh + base + idx4) = kh;
        tile[d + 0][s] = (_Float16)vv.x;
        tile[d + 1][s] = (_Float16)vv.y;
        tile[d + 2][s] = (_Float16)vv.z;
        tile[d + 3][s] = (_Float16)vv.w;
    }
    __syncthreads();
    #pragma unroll
    for (int e = 0; e < 2; ++e) {
        const int idx8 = (e * 256 + tid) * 8;
        const int d = idx8 >> 6, s = idx8 & 63;
        const v8h hv = *(const v8h*)&tile[d][s];
        int lo = __builtin_amdgcn_cvt_pk_fp8_f32((float)hv[0], (float)hv[1], 0,  false);
        lo     = __builtin_amdgcn_cvt_pk_fp8_f32((float)hv[2], (float)hv[3], lo, true);
        int hh = __builtin_amdgcn_cvt_pk_fp8_f32((float)hv[4], (float)hv[5], 0,  false);
        hh     = __builtin_amdgcn_cvt_pk_fp8_f32((float)hv[6], (float)hv[7], hh, true);
        v2i pk; pk[0] = lo; pk[1] = hh;
        *(v2i*)(V8T + ((size_t)b * DIM + d) * SEQ + s0 + s) = pk;
    }
}

#define N_TILE  128
#define KPAD    72
#define V8PAD   144

__global__ __launch_bounds__(NTHREADS)
void fa_kernel_f16(const float* __restrict__ Q, const _Float16* __restrict__ Kh,
                   const unsigned char* __restrict__ V8T, const float* __restrict__ MASK,
                   float* __restrict__ OUT)
{
    __shared__ __align__(128) _Float16      sK [2][N_TILE][KPAD];
    __shared__ __align__(128) unsigned char sV8[2][DIM][V8PAD];
    __shared__ __align__(128) float         sBias[2][N_TILE];

    const int tid  = threadIdx.x;
    const int wave = tid >> 5;
    const int lane = tid & 31;
    const int col  = lane & 15;
    const int hi   = lane >> 4;

    const int qtile = blockIdx.x;
    const int bb    = blockIdx.y;
    const size_t qbase_row = (size_t)bb * SEQ + (size_t)qtile * M_TILE + wave * 16;

    v16h bq0, bq1;
    {
        const float* qrow = Q + (qbase_row + col) * DIM;
        #pragma unroll
        for (int j = 0; j < 16; ++j) {
            bq0[j] = (_Float16)(qrow[hi * 16 + j]      * QS_L2E);
            bq1[j] = (_Float16)(qrow[32 + hi * 16 + j] * QS_L2E);
        }
    }
    v8i ones8;
    #pragma unroll
    for (int j = 0; j < 8; ++j) ones8[j] = 0x38383838;

    v8f acc[4], lfrag;
    #pragma unroll
    for (int r = 0; r < 8; ++r) lfrag[r] = 0.0f;
    #pragma unroll
    for (int g = 0; g < 4; ++g) {
        #pragma unroll
        for (int r = 0; r < 8; ++r) acc[g][r] = 0.0f;
    }

    const _Float16*      khBase = Kh  + (size_t)bb * SEQ * DIM;
    const unsigned char* v8Base = V8T + (size_t)bb * DIM * SEQ;
    const float*         mkBase = MASK + (size_t)bb * SEQ;

    auto STAGE = [&](int B, int KB) {
        const _Float16*      khT = khBase + (size_t)KB * DIM;
        const unsigned char* v8T = v8Base + KB;
        #pragma unroll
        for (int e = 0; e < (N_TILE * DIM) / (NTHREADS * 8); ++e) {
            const int idx8 = e * NTHREADS * 8 + tid * 8;
            const int ky = idx8 >> 6, dd = idx8 & 63;
            cp16(&sK[B][ky][dd], khT + idx8);
        }
        #pragma unroll
        for (int e = 0; e < (N_TILE * DIM) / (NTHREADS * 16); ++e) {
            const int idxB = e * NTHREADS * 16 + tid * 16;
            const int vd = idxB >> 7, vk = idxB & 127;
            cp16(&sV8[B][vd][vk], v8T + (size_t)vd * SEQ + vk);
        }
        sBias[B][tid] = (mkBase[KB + tid] > 0.0f) ? 0.0f : NEG_BIG;
        if (KB + N_TILE < SEQ) {
            __builtin_prefetch(khT + N_TILE * DIM + (size_t)tid * 64, 0, 3);
        }
    };

    STAGE(0, 0);
    int buf = 0;

    for (int kb = 0; kb < SEQ; kb += N_TILE) {
        async_wait0();
        __syncthreads();
        if (kb + N_TILE < SEQ) STAGE(buf ^ 1, kb + N_TILE);

        const _Float16      (*cK)[KPAD]  = sK[buf];
        const unsigned char (*cV)[V8PAD] = sV8[buf];
        const float*         cB          = sBias[buf];

        #pragma unroll
        for (int sc = 0; sc < 2; ++sc) {
            const int k0 = sc * 64;

            v8f st[4];
            #pragma unroll
            for (int kg = 0; kg < 4; ++kg)
                st[kg] = *(const v8f*)&cB[k0 + kg * 16 + hi * 8];
            #pragma unroll
            for (int c = 0; c < 2; ++c) {
                const int db = c * 32;
                const v16h bq = c ? bq1 : bq0;
                #pragma unroll
                for (int kg = 0; kg < 4; ++kg) {
                    const v16h a = cat8(*(const v8h*)&cK[k0 + kg * 16 + col][db + hi * 16],
                                        *(const v8h*)&cK[k0 + kg * 16 + col][db + hi * 16 + 8]);
                    st[kg] = __builtin_amdgcn_wmma_f32_16x16x32_f16(false, a, false, bq, (short)0, st[kg], false, false);
                }
            }

            v8i ap8;
            #pragma unroll
            for (int kg = 0; kg < 4; ++kg) {
                float p[8];
                #pragma unroll
                for (int r = 0; r < 8; ++r)
                    p[r] = __builtin_amdgcn_exp2f(st[kg][r]);
                int lo = __builtin_amdgcn_cvt_pk_fp8_f32(p[0], p[1], 0,  false);
                lo     = __builtin_amdgcn_cvt_pk_fp8_f32(p[2], p[3], lo, true);
                int hh = __builtin_amdgcn_cvt_pk_fp8_f32(p[4], p[5], 0,  false);
                hh     = __builtin_amdgcn_cvt_pk_fp8_f32(p[6], p[7], hh, true);
                ap8[2 * kg]     = lo;
                ap8[2 * kg + 1] = hh;
            }

            lfrag = __builtin_amdgcn_wmma_f32_16x16x64_fp8_fp8(ap8, ones8, (short)0, lfrag, false, false);

            #pragma unroll
            for (int g = 0; g < 4; ++g) {
                const v2i q0 = *(const v2i*)&cV[g * 16 + col][k0 + hi * 8];
                const v2i q1 = *(const v2i*)&cV[g * 16 + col][k0 + 16 + hi * 8];
                const v2i q2 = *(const v2i*)&cV[g * 16 + col][k0 + 32 + hi * 8];
                const v2i q3 = *(const v2i*)&cV[g * 16 + col][k0 + 48 + hi * 8];
                const v4ix b0 = __builtin_shufflevector(q0, q1, 0,1,2,3);
                const v4ix b1 = __builtin_shufflevector(q2, q3, 0,1,2,3);
                const v8i bv8 = __builtin_shufflevector(b0, b1, 0,1,2,3,4,5,6,7);
                acc[g] = __builtin_amdgcn_wmma_f32_16x16x64_fp8_fp8(ap8, bv8, (short)0, acc[g], false, false);
            }
        }
        buf ^= 1;
    }

    #pragma unroll
    for (int r = 0; r < 8; ++r) {
        const int row = r + hi * 8;
        const size_t qrow = qbase_row + row;
        const float mq   = MASK[qrow];
        const float invl = 1.0f / lfrag[r];
        #pragma unroll
        for (int g = 0; g < 4; ++g) {
            const int dd = g * 16 + col;
            const float o  = acc[g][r] * invl;
            const float qs = Q[qrow * DIM + dd] * QSCALE;
            OUT[qrow * DIM + dd] = o * mq + qs * (1.0f - mq);
        }
    }
}

#define FB_NT 64

__global__ __launch_bounds__(NTHREADS)
void fa_kernel_f32(const float* __restrict__ Q, const float* __restrict__ K,
                   const float* __restrict__ V, const float* __restrict__ MASK,
                   float* __restrict__ OUT)
{
    __shared__ __align__(128) _Float16 sK [FB_NT][KPAD];
    __shared__ __align__(128) _Float16 sVT[DIM][KPAD];
    __shared__ __align__(128) float    sBias[FB_NT];

    const int tid  = threadIdx.x;
    const int wave = tid >> 5;
    const int lane = tid & 31;
    const int col  = lane & 15;
    const int hi   = lane >> 4;

    const int qtile = blockIdx.x;
    const int bb    = blockIdx.y;
    const size_t qbase_row = (size_t)bb * SEQ + (size_t)qtile * M_TILE + wave * 16;

    v16h bq0, bq1;
    {
        const float* qrow = Q + (qbase_row + col) * DIM;
        #pragma unroll
        for (int j = 0; j < 16; ++j) {
            bq0[j] = (_Float16)(qrow[hi * 16 + j]      * QS_L2E);
            bq1[j] = (_Float16)(qrow[32 + hi * 16 + j] * QS_L2E);
        }
    }
    v16h bOnes;
    #pragma unroll
    for (int j = 0; j < 16; ++j) bOnes[j] = (_Float16)1.0f;

    v8f acc[4], lfrag;
    #pragma unroll
    for (int r = 0; r < 8; ++r) lfrag[r] = 0.0f;
    #pragma unroll
    for (int g = 0; g < 4; ++g) {
        #pragma unroll
        for (int r = 0; r < 8; ++r) acc[g][r] = 0.0f;
    }

    for (int kb = 0; kb < SEQ; kb += FB_NT) {
        const size_t tbase = (size_t)bb * SEQ + kb;
        #pragma unroll
        for (int e = 0; e < (FB_NT * DIM) / (NTHREADS * 4); ++e) {
            const int idx4 = (e * NTHREADS + tid) * 4;
            const int ky = idx4 >> 6;
            const int dd = idx4 & 63;
            const float4 kv = *(const float4*)(K + (tbase + ky) * DIM + dd);
            const float4 vv = *(const float4*)(V + (tbase + ky) * DIM + dd);
            v4h kh;
            kh[0] = (_Float16)kv.x; kh[1] = (_Float16)kv.y;
            kh[2] = (_Float16)kv.z; kh[3] = (_Float16)kv.w;
            *(v4h*)&sK[ky][dd] = kh;
            sVT[dd + 0][ky] = (_Float16)vv.x;
            sVT[dd + 1][ky] = (_Float16)vv.y;
            sVT[dd + 2][ky] = (_Float16)vv.z;
            sVT[dd + 3][ky] = (_Float16)vv.w;
        }
        if (tid < FB_NT) {
            const float mval = MASK[tbase + tid];
            sBias[tid] = (mval > 0.0f) ? 0.0f : NEG_BIG;
        }
        __syncthreads();

        #pragma unroll
        for (int ch = 0; ch < 2; ++ch) {
            const int k0 = ch * 32;
            v8f st0 = *(const v8f*)&sBias[k0 + hi * 8];
            v8f st1 = *(const v8f*)&sBias[k0 + 16 + hi * 8];
            #pragma unroll
            for (int c = 0; c < 2; ++c) {
                const int db = c * 32;
                const v16h a0 = cat8(*(const v8h*)&sK[k0 + col     ][db + hi * 16],
                                     *(const v8h*)&sK[k0 + col     ][db + hi * 16 + 8]);
                const v16h a1 = cat8(*(const v8h*)&sK[k0 + 16 + col][db + hi * 16],
                                     *(const v8h*)&sK[k0 + 16 + col][db + hi * 16 + 8]);
                const v16h bq = c ? bq1 : bq0;
                st0 = __builtin_amdgcn_wmma_f32_16x16x32_f16(false, a0, false, bq, (short)0, st0, false, false);
                st1 = __builtin_amdgcn_wmma_f32_16x16x32_f16(false, a1, false, bq, (short)0, st1, false, false);
            }
            v16h ap;
            #pragma unroll
            for (int r = 0; r < 8; ++r) {
                ap[r]     = (_Float16)__builtin_amdgcn_exp2f(st0[r]);
                ap[r + 8] = (_Float16)__builtin_amdgcn_exp2f(st1[r]);
            }
            lfrag = __builtin_amdgcn_wmma_f32_16x16x32_f16(false, ap, false, bOnes, (short)0, lfrag, false, false);
            #pragma unroll
            for (int g = 0; g < 4; ++g) {
                const v16h bv = cat8(*(const v8h*)&sVT[g * 16 + col][k0 + hi * 8],
                                     *(const v8h*)&sVT[g * 16 + col][k0 + 16 + hi * 8]);
                acc[g] = __builtin_amdgcn_wmma_f32_16x16x32_f16(false, ap, false, bv, (short)0, acc[g], false, false);
            }
        }
        __syncthreads();
    }

    #pragma unroll
    for (int r = 0; r < 8; ++r) {
        const int row = r + hi * 8;
        const size_t qrow = qbase_row + row;
        const float mq   = MASK[qrow];
        const float invl = 1.0f / lfrag[r];
        #pragma unroll
        for (int g = 0; g < 4; ++g) {
            const int dd = g * 16 + col;
            const float o  = acc[g][r] * invl;
            const float qs = Q[qrow * DIM + dd] * QSCALE;
            OUT[qrow * DIM + dd] = o * mq + qs * (1.0f - mq);
        }
    }
}

extern "C" void kernel_launch(void* const* d_in, const int* in_sizes, int n_in,
                              void* d_out, int out_size, void* d_ws, size_t ws_size,
                              hipStream_t stream) {
    (void)in_sizes; (void)n_in; (void)out_size;
    const float* q    = (const float*)d_in[0];
    const float* k    = (const float*)d_in[1];
    const float* v    = (const float*)d_in[2];
    const float* mask = (const float*)d_in[3];
    float* out = (float*)d_out;

    const size_t elems = (size_t)BATCH * SEQ * DIM;
    const size_t need  = elems * sizeof(_Float16) + elems;

    dim3 grid(SEQ / M_TILE, BATCH);
    dim3 block(NTHREADS);

    if (ws_size >= need) {
        _Float16*      Kh  = (_Float16*)d_ws;
        unsigned char* V8T = (unsigned char*)(Kh + elems);
        cvt_kernel<<<dim3(SEQ / 64, BATCH), dim3(256), 0, stream>>>(k, v, Kh, V8T);
        fa_kernel_f16<<<grid, block, 0, stream>>>(q, Kh, V8T, mask, out);
    } else {
        fa_kernel_f32<<<grid, block, 0, stream>>>(q, k, v, mask, out);
    }
}
